// ContextualLoss_63376537419923
// MI455X (gfx1250) — matrix operand order repaired and checked by reading, not compiled, not run on hardware
//
#include <hip/hip_runtime.h>
#include <hip/hip_bf16.h>
#include <math.h>

// Problem constants (fixed by reference: X,Y = (4, 256, 64, 64) fp32)
#define BB 4
#define CC 256
#define NN 4096           // 64*64 spatial positions
#define EPS_NORM 2.220446049250313e-16f
#define EPS_MIN  1e-5f
// 1/H_BANDWIDTH = 10

typedef _Float16 v16h __attribute__((ext_vector_type(16)));
typedef _Float16 v8h  __attribute__((ext_vector_type(8)));
typedef float    v8f  __attribute__((ext_vector_type(8)));
typedef unsigned u32x4 __attribute__((ext_vector_type(4)));
typedef int      i32x8 __attribute__((ext_vector_type(8)));
typedef int      i32x4 __attribute__((ext_vector_type(4)));

// ---- workspace layout (bytes), all 256B aligned ----
#define OFF_YMEAN  0u                       // B*C floats        = 4096 B
#define OFF_INVNX  4096u                    // B*N floats        = 65536 B
#define OFF_INVNY  (4096u+65536u)           // B*N floats
#define OFF_SMAX   (4096u+2u*65536u)        // B*N u32 keys (row max of S)
#define OFF_RSUM   (4096u+3u*65536u)        // B*N floats  (row sum of w)
#define OFF_CMAX   (4096u+4u*65536u)        // B*N u32 keys (col max of A)
#define OFF_XH     (4096u+5u*65536u)        // B*N*C halfs = 8 MB
#define OFF_YH     (OFF_XH + 2u*BB*NN*CC)   // B*N*C halfs = 8 MB

// Monotone float<->u32 key so unsigned atomicMax implements float max.
__device__ __forceinline__ unsigned fkey(float f) {
    unsigned u = __float_as_uint(f);
    return (u & 0x80000000u) ? ~u : (u | 0x80000000u);
}
__device__ __forceinline__ float funkey(unsigned u) {
    return (u & 0x80000000u) ? __uint_as_float(u & 0x7FFFFFFFu)
                             : __uint_as_float(~u);
}

// ---- Tensor Data Mover: 2D tile (32 halfs wide x `rows` tall) global->LDS.
// Tensor row stride = CC elements. LDS padding: after every 16 DWORDs (64B
// row) insert 4 DWORDs (16B) -> LDS row stride = 80B = 40 halfs, matching the
// padded fragment-gather layout.
__device__ __forceinline__ void tdm_load_tile(const _Float16* gaddr,
                                              unsigned lds_off, unsigned rows) {
    unsigned long long ga = (unsigned long long)(size_t)gaddr;
    u32x4 g0;
    g0.x = 1u;                                        // count=1, user desc
    g0.y = lds_off;                                   // lds_addr [63:32]
    g0.z = (unsigned)(ga & 0xFFFFFFFFull);            // global_addr lo
    g0.w = (unsigned)((ga >> 32) & 0x01FFFFFFull)     // global_addr hi (56:32)
         | (2u << 30);                                // type=2 ("image")
    i32x8 g1;
    g1[0] = (int)((1u << 16)      // data_size = 1 -> 2-byte elements
                | (1u << 20)      // pad_enable
                | (3u << 22)      // pad_interval: 16 DWORDs (64B) per row
                | (3u << 25));    // pad_amount: 4 DWORDs (16B)
    g1[1] = (int)(((unsigned)CC & 0xFFFFu) << 16);    // tensor_dim0 lo16
    g1[2] = (int)((((unsigned)CC >> 16) & 0xFFFFu)    // tensor_dim0 hi16
                | (((unsigned)NN & 0xFFFFu) << 16));  // tensor_dim1 lo16
    g1[3] = (int)((((unsigned)NN >> 16) & 0xFFFFu)    // tensor_dim1 hi16
                | (32u << 16));                       // tile_dim0 = 32 elems
    g1[4] = (int)rows;                                // tile_dim1
    g1[5] = (int)CC;                                  // tensor_dim0_stride lo
    g1[6] = 0; g1[7] = 0;
    i32x4 gz = {0, 0, 0, 0};
#if __has_include(<hip/amd_detail/amd_gfx1250_TDM.h>)
    i32x8 gz8 = {0, 0, 0, 0, 0, 0, 0, 0};
    __builtin_amdgcn_tensor_load_to_lds(g0, g1, gz, gz, gz8, 0);
#else
    __builtin_amdgcn_tensor_load_to_lds(g0, g1, gz, gz, 0);
#endif
}

// ---------------- zero the atomic reduction buffers (every call) -------------
__global__ void init_kernel(unsigned* p, int n) {
    int i = blockIdx.x * blockDim.x + threadIdx.x;
    if (i < n) p[i] = 0u;   // 0 is below every fkey and is float 0.0 for rsum
}

// ---------------- per-(b,c) spatial mean of Y --------------------------------
__global__ __launch_bounds__(256) void ymean_kernel(const float* __restrict__ Y,
                                                    float* __restrict__ ymean) {
    __shared__ float sd[256];
    int bc = blockIdx.x;                 // b*C + c
    const float* p = Y + (size_t)bc * NN;
    float s = 0.f;
    #pragma unroll
    for (int k = 0; k < 16; ++k) s += p[threadIdx.x + k * 256];
    sd[threadIdx.x] = s;
    __syncthreads();
    for (int o = 128; o > 0; o >>= 1) {
        if (threadIdx.x < o) sd[threadIdx.x] += sd[threadIdx.x + o];
        __syncthreads();
    }
    if (threadIdx.x == 0) ymean[bc] = sd[0] * (1.0f / NN);
}

// ---------------- per-(b,n) inverse channel L2 norms of centered X,Y ---------
__global__ __launch_bounds__(256) void norm_kernel(const float* __restrict__ X,
                                                   const float* __restrict__ Y,
                                                   const float* __restrict__ ymean,
                                                   float* __restrict__ invnx,
                                                   float* __restrict__ invny) {
    __shared__ float sym[CC];
    int gid = blockIdx.x * 256 + threadIdx.x;   // over B*N, one b per block
    int b = gid >> 12, n = gid & (NN - 1);
    sym[threadIdx.x] = ymean[b * CC + threadIdx.x];
    __syncthreads();
    float sx = 0.f, sy = 0.f;
    for (int c = 0; c < CC; ++c) {
        size_t off = ((size_t)(b * CC + c)) * NN + n;
        float ym = sym[c];
        float x = X[off] - ym; sx += x * x;
        float y = Y[off] - ym; sy += y * y;
    }
    invnx[gid] = 1.0f / (sqrtf(sx) + EPS_NORM);
    invny[gid] = 1.0f / (sqrtf(sy) + EPS_NORM);
}

// ------- transpose [B,C,N] -> [B,N,C], center, scale, cast to f16 ------------
__global__ __launch_bounds__(256) void tohalf_kernel(const float* __restrict__ X,
                                                     const float* __restrict__ Y,
                                                     const float* __restrict__ ymean,
                                                     const float* __restrict__ invnx,
                                                     const float* __restrict__ invny,
                                                     _Float16* __restrict__ Xh,
                                                     _Float16* __restrict__ Yh) {
    __shared__ float tile[32][65];
    __shared__ float s_ym[32];
    __shared__ float s_inv[64];
    int b = blockIdx.z, c0 = blockIdx.y * 32, n0 = blockIdx.x * 64;
    int tid = threadIdx.x;
    if (tid < 32) s_ym[tid] = ymean[b * CC + c0 + tid];

    for (int s = 0; s < 2; ++s) {
        const float* src = s ? Y : X;
        const float* inv = s ? invny : invnx;
        _Float16* dst    = s ? Yh : Xh;
        if (tid < 64) s_inv[tid] = inv[b * NN + n0 + tid];
        __syncthreads();
        #pragma unroll
        for (int r = 0; r < 8; ++r) {
            int cl = r * 4 + (tid >> 6), nl = tid & 63;
            tile[cl][nl] = src[((size_t)(b * CC + c0 + cl)) * NN + n0 + nl] - s_ym[cl];
        }
        __syncthreads();
        #pragma unroll
        for (int r = 0; r < 8; ++r) {
            int nl = r * 8 + (tid >> 5), cl = tid & 31;
            dst[((size_t)(b * NN + n0 + nl)) * CC + c0 + cl] =
                (_Float16)(tile[cl][nl] * s_inv[nl]);
        }
        __syncthreads();
    }
}

// ---------------- 3-pass fused GEMM + reductions -----------------------------
// Block tile 128(i) x 64(j); 8 waves as 4(i) x 2(j); each wave 32x32 (4 WMMA
// per K-step). A/B tiles DMA'd into double-buffered LDS by the TDM, issued by
// wave 0 one K-step ahead (TENSORcnt in-order per wave => wait<=2 releases the
// current tiles while the next pair streams in).
// mode 0: rowmax_j S[i,j]  -> smaxKey      (gives d_min = 1 - smax)
// mode 1: rowsum_j exp(ci + ki*S) -> rsum
// mode 2: colmax_i exp(ci + ki*S)/r_i -> cmaxKey
__global__ __launch_bounds__(256) void gemm_pass(const _Float16* __restrict__ Xh,
                                                 const _Float16* __restrict__ Yh,
                                                 unsigned* __restrict__ smaxKey,
                                                 float* __restrict__ rsum,
                                                 unsigned* __restrict__ cmaxKey,
                                                 int mode) {
    __shared__ __align__(16) _Float16 As[2][128 * 40];  // padded K stride 40
    __shared__ __align__(16) _Float16 Bs[2][64 * 40];

    int tid = threadIdx.x;
    int lane = tid & 31, w = tid >> 5;
    int wi = w & 3, wj = w >> 2;                 // 4 (i) x 2 (j) waves
    int b = blockIdx.z, i0 = blockIdx.y * 128, j0 = blockIdx.x * 64;

    const _Float16* Xb = Xh + ((size_t)(b * NN + i0)) * CC;
    const _Float16* Yb = Yh + ((size_t)(b * NN + j0)) * CC;

    unsigned ldsA0 = (unsigned)(size_t)(void*)&As[0][0];
    unsigned ldsA1 = (unsigned)(size_t)(void*)&As[1][0];
    unsigned ldsB0 = (unsigned)(size_t)(void*)&Bs[0][0];
    unsigned ldsB1 = (unsigned)(size_t)(void*)&Bs[1][0];

    v8f acc[2][2] = {};

    if (w == 0) {                       // prime tile 0
        tdm_load_tile(Xb, ldsA0, 128);
        tdm_load_tile(Yb, ldsB0, 64);
    }

    #pragma unroll
    for (int step = 0; step < CC / 32; ++step) {
        int cur = step & 1;
        if (w == 0) {
            if (step + 1 < CC / 32) {   // stream next tiles into other buffer
                int k1 = (step + 1) * 32;
                tdm_load_tile(Xb + k1, cur ? ldsA0 : ldsA1, 128);
                tdm_load_tile(Yb + k1, cur ? ldsB0 : ldsB1, 64);
                __builtin_amdgcn_s_wait_tensorcnt(2);  // current pair resident
            } else {
                __builtin_amdgcn_s_wait_tensorcnt(0);
            }
        }
        __syncthreads();

        const _Float16* Ac = As[cur];
        const _Float16* Bc = Bs[cur];
        int half = lane >> 4, m = lane & 15;
        int kbA = half * 8, kbB = half * 8;

        v16h a[2], fb[2];
        #pragma unroll
        for (int f = 0; f < 2; ++f) {   // A frags: M = m, K {kb..+7}U{kb+16..+23}
            int arow = wi * 32 + f * 16 + m;
            v8h lo = *(const v8h*)&Ac[arow * 40 + kbA];
            v8h hi = *(const v8h*)&Ac[arow * 40 + kbA + 16];
            #pragma unroll
            for (int t = 0; t < 8; ++t) { a[f][t] = lo[t]; a[f][8 + t] = hi[t]; }
        }
        #pragma unroll
        for (int g = 0; g < 2; ++g) {   // B frags: N = m, K {kb..+7}U{kb+16..+23}
            int bcol = wj * 32 + g * 16 + m;
            v8h lo = *(const v8h*)&Bc[bcol * 40 + kbB];
            v8h hi = *(const v8h*)&Bc[bcol * 40 + kbB + 16];
            #pragma unroll
            for (int t = 0; t < 8; ++t) { fb[g][t] = lo[t]; fb[g][8 + t] = hi[t]; }
        }

        acc[0][0] = __builtin_amdgcn_wmma_f32_16x16x32_f16(false, a[0], false, fb[0],
                                                           (short)0, acc[0][0], false, false);
        acc[0][1] = __builtin_amdgcn_wmma_f32_16x16x32_f16(false, a[0], false, fb[1],
                                                           (short)0, acc[0][1], false, false);
        acc[1][0] = __builtin_amdgcn_wmma_f32_16x16x32_f16(false, a[1], false, fb[0],
                                                           (short)0, acc[1][0], false, false);
        acc[1][1] = __builtin_amdgcn_wmma_f32_16x16x32_f16(false, a[1], false, fb[1],
                                                           (short)0, acc[1][1], false, false);
        __syncthreads();                // buffer free for the next TDM issue
    }

    // C/D layout: VGPR v, lanes 0-15 -> M=v, lanes 16-31 -> M=v+8; N = lane%16
    int halfsel = lane >> 4, nloc = lane & 15;

    if (mode == 0) {
        #pragma unroll
        for (int f = 0; f < 2; ++f) {
            int rowbase = i0 + wi * 32 + f * 16 + halfsel * 8;
            #pragma unroll
            for (int v = 0; v < 8; ++v) {
                float val = fmaxf(acc[f][0][v], acc[f][1][v]);
                val = fmaxf(val, __shfl_xor(val, 1, 32));
                val = fmaxf(val, __shfl_xor(val, 2, 32));
                val = fmaxf(val, __shfl_xor(val, 4, 32));
                val = fmaxf(val, __shfl_xor(val, 8, 32));
                if (nloc == 0)
                    atomicMax(&smaxKey[b * NN + rowbase + v], fkey(val));
            }
        }
    } else if (mode == 1) {
        #pragma unroll
        for (int f = 0; f < 2; ++f) {
            int rowbase = i0 + wi * 32 + f * 16 + halfsel * 8;
            #pragma unroll
            for (int v = 0; v < 8; ++v) {
                int row = b * NN + rowbase + v;
                float smax = funkey(smaxKey[row]);
                float Dm = (1.0f - smax) + EPS_MIN;   // min_j d + eps
                float ki = 10.0f / Dm;
                float ci = 10.0f - ki;
                float s = __expf(ci + ki * acc[f][0][v]) +
                          __expf(ci + ki * acc[f][1][v]);
                s += __shfl_xor(s, 1, 32);
                s += __shfl_xor(s, 2, 32);
                s += __shfl_xor(s, 4, 32);
                s += __shfl_xor(s, 8, 32);
                if (nloc == 0) atomicAdd(&rsum[row], s);
            }
        }
    } else {
        float cm0 = 0.f, cm1 = 0.f;
        #pragma unroll
        for (int f = 0; f < 2; ++f) {
            int rowbase = i0 + wi * 32 + f * 16 + halfsel * 8;
            #pragma unroll
            for (int v = 0; v < 8; ++v) {
                int row = b * NN + rowbase + v;
                float smax = funkey(smaxKey[row]);
                float Dm = (1.0f - smax) + EPS_MIN;
                float ki = 10.0f / Dm;
                float ci = 10.0f - ki;
                float rinv = 1.0f / rsum[row];
                cm0 = fmaxf(cm0, __expf(ci + ki * acc[f][0][v]) * rinv);
                cm1 = fmaxf(cm1, __expf(ci + ki * acc[f][1][v]) * rinv);
            }
        }
        cm0 = fmaxf(cm0, __shfl_xor(cm0, 16, 32));
        cm1 = fmaxf(cm1, __shfl_xor(cm1, 16, 32));
        if (lane < 16) {
            atomicMax(&cmaxKey[b * NN + j0 + wj * 32 + lane],      fkey(cm0));
            atomicMax(&cmaxKey[b * NN + j0 + wj * 32 + 16 + lane], fkey(cm1));
        }
    }
}

// ---------------- CX[b] = mean_j colmax; out = -log(CX) ----------------------
__global__ __launch_bounds__(256) void final_kernel(const unsigned* __restrict__ cmaxKey,
                                                    float* __restrict__ out) {
    __shared__ float sd[256];
    int b = blockIdx.x;
    float s = 0.f;
    #pragma unroll
    for (int k = 0; k < 16; ++k)
        s += funkey(cmaxKey[b * NN + threadIdx.x + k * 256]);
    sd[threadIdx.x] = s;
    __syncthreads();
    for (int o = 128; o > 0; o >>= 1) {
        if (threadIdx.x < o) sd[threadIdx.x] += sd[threadIdx.x + o];
        __syncthreads();
    }
    if (threadIdx.x == 0) out[b] = -logf(sd[0] * (1.0f / NN));
}

extern "C" void kernel_launch(void* const* d_in, const int* in_sizes, int n_in,
                              void* d_out, int out_size, void* d_ws, size_t ws_size,
                              hipStream_t stream) {
    const float* X = (const float*)d_in[0];
    const float* Y = (const float*)d_in[1];
    float* out = (float*)d_out;

    char* ws = (char*)d_ws;
    float*    ymean   = (float*)(ws + OFF_YMEAN);
    float*    invnx   = (float*)(ws + OFF_INVNX);
    float*    invny   = (float*)(ws + OFF_INVNY);
    unsigned* smaxKey = (unsigned*)(ws + OFF_SMAX);
    float*    rsum    = (float*)(ws + OFF_RSUM);
    unsigned* cmaxKey = (unsigned*)(ws + OFF_CMAX);
    _Float16* Xh      = (_Float16*)(ws + OFF_XH);
    _Float16* Yh      = (_Float16*)(ws + OFF_YH);

    // smaxKey, rsum, cmaxKey are contiguous: zero all 3*B*N words each call
    init_kernel<<<(3 * BB * NN + 255) / 256, 256, 0, stream>>>(smaxKey, 3 * BB * NN);
    ymean_kernel<<<BB * CC, 256, 0, stream>>>(Y, ymean);
    norm_kernel<<<(BB * NN) / 256, 256, 0, stream>>>(X, Y, ymean, invnx, invny);
    tohalf_kernel<<<dim3(NN / 64, CC / 32, BB), 256, 0, stream>>>(X, Y, ymean,
                                                                  invnx, invny, Xh, Yh);
    dim3 g(NN / 64, NN / 128, BB);
    gemm_pass<<<g, 256, 0, stream>>>(Xh, Yh, smaxKey, rsum, cmaxKey, 0);
    gemm_pass<<<g, 256, 0, stream>>>(Xh, Yh, smaxKey, rsum, cmaxKey, 1);
    gemm_pass<<<g, 256, 0, stream>>>(Xh, Yh, smaxKey, rsum, cmaxKey, 2);
    final_kernel<<<BB, 256, 0, stream>>>(cmaxKey, out);
}
